// MultiHeadAttention_78202764525783
// MI455X (gfx1250) — matrix operand order repaired and checked by reading, not compiled, not run on hardware
//
#include <hip/hip_runtime.h>

// ---------------------------------------------------------------------------
// MultiHeadAttention forward for MI455X (gfx1250, wave32, WMMA)
// B=2, S=2048, D=512, H=8, DH=64.  Outputs: attn [B,H,S,S] f32 ++ final [B,S,D] f32
// ---------------------------------------------------------------------------

typedef __attribute__((ext_vector_type(16))) _Float16 v16h;
typedef __attribute__((ext_vector_type(8)))  float    v8f;

static constexpr int Bk = 2, Sk = 2048, Dk = 512, Hh = 8, DH = 64;

__device__ __forceinline__ v8f wmma_f16(v16h a, v16h b, v8f c) {
  // v_wmma_f32_16x16x32_f16  (D = A x B + C, f32 accumulate)
  return __builtin_amdgcn_wmma_f32_16x16x32_f16(
      /*neg_a=*/false, a, /*neg_b=*/false, b,
      /*c_mod=*/(short)0, c, /*reuse_a=*/false, /*reuse_b=*/false);
}

// A-fragment (16x32 f16, MxK): lane m=lane&15 holds row m; half-wave selects K
// sub-bands {0-7,16-23} (lanes 0-15) / {8-15,24-31} (lanes 16-31).
__device__ __forceinline__ v16h a_frag_f16(const _Float16* base, int stride, int k0) {
  const int lane = threadIdx.x & 31;
  const int m    = lane & 15;
  const int koff = k0 + ((lane >> 4) << 3);
  union { v16h v; _Float16 h[16]; } u;
#pragma unroll
  for (int e = 0; e < 16; ++e)
    u.h[e] = base[m * stride + koff + (e & 7) + ((e >> 3) << 4)];
  return u.v;
}

// Same A layout but source is f32 (converted on the fly) — used for attn probs.
__device__ __forceinline__ v16h a_frag_f32(const float* base, int stride, int k0) {
  const int lane = threadIdx.x & 31;
  const int m    = lane & 15;
  const int koff = k0 + ((lane >> 4) << 3);
  union { v16h v; _Float16 h[16]; } u;
#pragma unroll
  for (int e = 0; e < 16; ++e)
    u.h[e] = (_Float16)base[m * stride + koff + (e & 7) + ((e >> 3) << 4)];
  return u.v;
}

// B-fragment (32x16 f16, KxN): lane holds column N=lane&15; K order matches A frag.
// Here B = K-matrix transposed: B[k_depth][n_key] = Ktile[n_key][k_depth].
__device__ __forceinline__ v16h b_frag_kT(const _Float16* ktile, int k0) {
  const int lane = threadIdx.x & 31;
  union { v16h v; _Float16 h[16]; } u;
#pragma unroll
  for (int e = 0; e < 16; ++e)
    u.h[e] = ktile[(lane & 15) * DH + k0 + ((lane >> 4) << 3) + (e & 7) + ((e >> 3) << 4)];
  return u.v;
}

// ---------------------------------------------------------------------------
// K0: Wo (f32) -> f16
// ---------------------------------------------------------------------------
__global__ void __launch_bounds__(256)
cvt_wo_kernel(const float* __restrict__ src, _Float16* __restrict__ dst, int n) {
  int i = blockIdx.x * 256 + threadIdx.x;
  if (i < n) dst[i] = (_Float16)src[i];
}

// ---------------------------------------------------------------------------
// K1: Q = x*Wq+bq, K = x*Wk+bk  (f16 out, head-major [B,H,S,DH])
// grid.x = (B*S)/16 = 256 row-tiles, 256 threads (8 waves).
// ---------------------------------------------------------------------------
__global__ void __launch_bounds__(256)
qk_proj_kernel(const float* __restrict__ x,
               const float* __restrict__ Wq, const float* __restrict__ bq,
               const float* __restrict__ Wk, const float* __restrict__ bk,
               _Float16* __restrict__ qbuf, _Float16* __restrict__ kbuf) {
  __shared__ _Float16 xt[16 * Dk];           // 16 KB: row tile of x in f16
  const int rt = blockIdx.x;
  for (int i = threadIdx.x; i < 16 * Dk; i += 256) {
    int r = i >> 9, c = i & (Dk - 1);
    xt[i] = (_Float16)x[(size_t)(rt * 16 + r) * Dk + c];
  }
  __syncthreads();

  const int lane = threadIdx.x & 31, wave = threadIdx.x >> 5;
  for (int t = wave; t < 64; t += 8) {       // 32 col-tiles x {Wq,Wk}
    const float* W    = (t < 32) ? Wq : Wk;
    const float* bias = (t < 32) ? bq : bk;
    _Float16*    dst  = (t < 32) ? qbuf : kbuf;
    const int ct = t & 31;
    v8f acc = {};
#pragma unroll 4
    for (int ch = 0; ch < 16; ++ch) {        // K = 512 in chunks of 32
      const int k0 = ch * 32;
      v16h a = a_frag_f16(xt, Dk, k0);
      union { v16h v; _Float16 h[16]; } ub;
      const float* wc = W + ct * 16 + (lane & 15);
#pragma unroll
      for (int e = 0; e < 16; ++e)
        ub.h[e] = (_Float16)wc[(size_t)(k0 + ((lane >> 4) << 3) + (e & 7) + ((e >> 3) << 4)) * Dk];
      acc = wmma_f16(a, ub.v, acc);
    }
    const int n  = lane & 15;
    const float bn = bias[ct * 16 + n];
#pragma unroll
    for (int r = 0; r < 8; ++r) {
      int m    = r + ((lane >> 4) << 3);
      int grow = rt * 16 + m;                // 0..B*S-1
      int b    = grow >> 11, s = grow & (Sk - 1);
      int col  = ct * 16 + n;
      int h    = col >> 6, d = col & (DH - 1);
      dst[(((size_t)(b * Hh + h) * Sk + s) << 6) + d] = (_Float16)(acc[r] + bn);
    }
  }
}

// ---------------------------------------------------------------------------
// K2: fused scores -> softmax -> attn write -> attn*V, one WG per (b,h,qtile).
// grid.x = B*H*(S/16) = 2048, 256 threads, ~151 KB dynamic LDS.
// ---------------------------------------------------------------------------
static constexpr size_t SC_B   = 16 * Sk * 4;         // 131072
static constexpr size_t QT_B   = 16 * DH * 2;         //   2048
static constexpr size_t KS_B   = 8 * 16 * DH * 2;     //  16384
static constexpr size_t RED_B  = 256 * 4;             //   1024
static constexpr size_t ORED_B = 4 * 256 * 4;         //   4096
static constexpr size_t SMEM_K2 = SC_B + QT_B + KS_B + RED_B + ORED_B; // 154624

__global__ void __launch_bounds__(256)
attn_kernel(const _Float16* __restrict__ qbuf, const _Float16* __restrict__ kbuf,
            const float* __restrict__ mask,
            float* __restrict__ attnOut, _Float16* __restrict__ outf16) {
  extern __shared__ char smem[];
  float*    sc     = (float*)smem;                         // 16 x 2048 scores
  _Float16* qtile  = (_Float16*)(smem + SC_B);             // 16 x 64 (pre-scaled)
  _Float16* kstage = (_Float16*)(smem + SC_B + QT_B);      // 8 waves x 16 x 64
  float*    red    = (float*)(smem + SC_B + QT_B + KS_B);  // softmax reductions
  float*    outred = (float*)(smem + SC_B + QT_B + KS_B + RED_B);

  const int bid  = blockIdx.x;
  const int qt   = bid & 127;
  const int h    = (bid >> 7) & 7;
  const int b    = bid >> 10;
  const int lane = threadIdx.x & 31, wave = threadIdx.x >> 5;
  const int qbase = qt * 16;

  const _Float16* qh = qbuf + (size_t)(b * Hh + h) * Sk * DH;
  const _Float16* kh = kbuf + (size_t)(b * Hh + h) * Sk * DH;

  // ---- stage Q tile, pre-scaled by 1/sqrt(DH) = 1/8 (exact in f16) ----
  for (int i = threadIdx.x; i < 16 * DH; i += 256)
    qtile[i] = qh[(size_t)(qbase + (i >> 6)) * DH + (i & 63)] * (_Float16)0.125f;
  __syncthreads();

  const v16h aq0 = a_frag_f16(qtile, DH, 0);
  const v16h aq1 = a_frag_f16(qtile, DH, 32);

  // Wave-relative LDS byte address of this wave's K staging buffer.
  // Dynamic LDS begins right after static LDS (none here): groupstaticsize().
  const uint32_t myk_lds_base =
      (uint32_t)__builtin_amdgcn_groupstaticsize() +
      (uint32_t)(SC_B + QT_B) + (uint32_t)wave * (16u * DH * 2u);
  _Float16* myk = kstage + wave * (16 * DH);

  // ---- phase 2: scores = (Q/8) K^T + mask*(-1e9) into LDS ----
  for (int it = 0; it < 16; ++it) {          // 128 key tiles / 8 waves
    const int kt = it * 8 + wave;
    const int kb = kt * 16;
    {
      // Async memory->LDS copy of the 16x64 f16 K tile (2 KB):
      // 4 x (32 lanes x 16 B) GLOBAL_LOAD_ASYNC_TO_LDS_B128, ASYNCcnt-tracked.
      const uint64_t gsrc = (uint64_t)(uintptr_t)(kh + (size_t)kb * DH);
#pragma unroll
      for (int p = 0; p < 4; ++p) {
        const uint32_t boff = (uint32_t)(p * 32 + lane) * 16u;
        const uint32_t ldst = myk_lds_base + boff;
        const uint64_t ga   = gsrc + boff;
        asm volatile("global_load_async_to_lds_b128 %0, %1, off"
                     :: "v"(ldst), "v"(ga) : "memory");
      }
      asm volatile("s_wait_asynccnt 0" ::: "memory");
    }
    __syncthreads();
    v8f acc = {};
    acc = wmma_f16(aq0, b_frag_kT(myk, 0),  acc);
    acc = wmma_f16(aq1, b_frag_kT(myk, 32), acc);
    const int n   = lane & 15;
    const int col = kb + n;
    const float mv = -1.0e9f * mask[b * Sk + col];
#pragma unroll
    for (int r = 0; r < 8; ++r) {
      int m = r + ((lane >> 4) << 3);
      sc[m * Sk + col] = acc[r] + mv;
    }
  }
  __syncthreads();

  // ---- phase 3: row softmax (16 rows x 16 threads) + write attn ----
  {
    const int row = threadIdx.x >> 4;
    const int l16 = threadIdx.x & 15;
    float* srow = sc + row * Sk;
    float mx = -3.0e38f;
    for (int c = l16; c < Sk; c += 16) mx = fmaxf(mx, srow[c]);
    red[threadIdx.x] = mx;
    __syncthreads();
    for (int s = 8; s > 0; s >>= 1) {
      if (l16 < s) red[row * 16 + l16] = fmaxf(red[row * 16 + l16], red[row * 16 + l16 + s]);
      __syncthreads();
    }
    const float rowmax = red[row * 16];
    __syncthreads();
    float sum = 0.f;
    for (int c = l16; c < Sk; c += 16) {
      float e = __expf(srow[c] - rowmax);
      srow[c] = e;
      sum += e;
    }
    red[threadIdx.x] = sum;
    __syncthreads();
    for (int s = 8; s > 0; s >>= 1) {
      if (l16 < s) red[row * 16 + l16] += red[row * 16 + l16 + s];
      __syncthreads();
    }
    const float inv = 1.0f / red[row * 16];
    float* arow = attnOut + ((size_t)(b * Hh + h) * Sk + qbase + row) * Sk;
    for (int c = l16; c < Sk; c += 16) {
      float p = srow[c] * inv;
      srow[c] = p;
      arow[c] = p;                            // 128 KB/WG: the HBM-bound part
    }
  }
  __syncthreads();

  // ---- phase 4: out(16x64) = attn(16x2048) * V(2048x64), V = Q (source bug) ----
  {
    const int ct   = wave & 3;                // 16-col tile of DH
    const int half = wave >> 2;               // split keys 2-way
    v8f acc = {};
#pragma unroll 2
    for (int ch = 0; ch < 32; ++ch) {
      const int kb = half * 1024 + ch * 32;
      v16h a = a_frag_f32(sc, Sk, kb);
      union { v16h v; _Float16 h[16]; } ub;   // B cols: V[key][n], n = lane&15, keys in A-frag order
      const _Float16* vr = qh + (size_t)kb * DH + ct * 16 + (lane & 15);
#pragma unroll
      for (int e = 0; e < 16; ++e)
        ub.h[e] = vr[(size_t)(((lane >> 4) << 3) + (e & 7) + ((e >> 3) << 4)) * DH];
      acc = wmma_f16(a, ub.v, acc);
    }
    if (half == 1) {
#pragma unroll
      for (int r = 0; r < 8; ++r) outred[ct * 256 + lane * 8 + r] = acc[r];
    }
    __syncthreads();
    if (half == 0) {
      const int n = lane & 15;
#pragma unroll
      for (int r = 0; r < 8; ++r) {
        int m = r + ((lane >> 4) << 3);
        float v = acc[r] + outred[ct * 256 + lane * 8 + r];
        outf16[(size_t)(b * Sk + qbase + m) * Dk + h * DH + ct * 16 + n] = (_Float16)v;
      }
    }
  }
}

// ---------------------------------------------------------------------------
// K3: final = out_f16 * Wo_f16 + bo   (f32 out)
// ---------------------------------------------------------------------------
__global__ void __launch_bounds__(256)
out_proj_kernel(const _Float16* __restrict__ outf16, const _Float16* __restrict__ WoH,
                const float* __restrict__ bo, float* __restrict__ finalOut) {
  __shared__ _Float16 at[16 * Dk];           // 16 KB row tile
  const int rt = blockIdx.x;
  {
    const uint4* src = (const uint4*)(outf16 + (size_t)rt * 16 * Dk);
    uint4*       dst = (uint4*)at;
    for (int i = threadIdx.x; i < (16 * Dk) / 8; i += 256) dst[i] = src[i];
  }
  __syncthreads();

  const int lane = threadIdx.x & 31, wave = threadIdx.x >> 5;
  for (int ct = wave; ct < 32; ct += 8) {
    v8f acc = {};
#pragma unroll 4
    for (int ch = 0; ch < 16; ++ch) {
      const int k0 = ch * 32;
      v16h a = a_frag_f16(at, Dk, k0);
      union { v16h v; _Float16 h[16]; } ub;
      const _Float16* wr = WoH + (size_t)k0 * Dk + ct * 16 + (lane & 15);
#pragma unroll
      for (int e = 0; e < 16; ++e)
        ub.h[e] = wr[(size_t)(((lane >> 4) << 3) + (e & 7) + ((e >> 3) << 4)) * Dk];
      acc = wmma_f16(a, ub.v, acc);
    }
    const int n  = lane & 15;
    const float bn = bo[ct * 16 + n];
#pragma unroll
    for (int r = 0; r < 8; ++r) {
      int m = r + ((lane >> 4) << 3);
      finalOut[(size_t)(rt * 16 + m) * Dk + ct * 16 + n] = acc[r] + bn;
    }
  }
}

// ---------------------------------------------------------------------------
extern "C" void kernel_launch(void* const* d_in, const int* in_sizes, int n_in,
                              void* d_out, int out_size, void* d_ws, size_t ws_size,
                              hipStream_t stream) {
  const float* x    = (const float*)d_in[0];
  const float* mask = (const float*)d_in[1];
  const float* Wq   = (const float*)d_in[2];
  const float* bq   = (const float*)d_in[3];
  const float* Wk   = (const float*)d_in[4];
  const float* bk   = (const float*)d_in[5];
  // d_in[6]=Wv, d_in[7]=bv are unused (reference bug: V uses Wq/bq)
  const float* Wo   = (const float*)d_in[8];
  const float* bo   = (const float*)d_in[9];

  // Workspace carve (f16): qbuf, kbuf [B,H,S,DH]; outf16 [B*S,D]; WoH [D,D]
  const size_t nQK = (size_t)Bk * Hh * Sk * DH;        // 2,097,152
  _Float16* qbuf   = (_Float16*)d_ws;
  _Float16* kbuf   = qbuf + nQK;
  _Float16* outf16 = kbuf + nQK;
  _Float16* WoH    = outf16 + (size_t)Bk * Sk * Dk;    // total ~12.6 MB

  float* attnOut  = (float*)d_out;                     // [B,H,S,S]
  float* finalOut = attnOut + (size_t)Bk * Hh * Sk * Sk;

  cvt_wo_kernel<<<(Dk * Dk + 255) / 256, 256, 0, stream>>>(Wo, WoH, Dk * Dk);
  qk_proj_kernel<<<(Bk * Sk) / 16, 256, 0, stream>>>(x, Wq, bq, Wk, bk, qbuf, kbuf);
  attn_kernel<<<Bk * Hh * (Sk / 16), 256, SMEM_K2, stream>>>(qbuf, kbuf, mask,
                                                             attnOut, outf16);
  out_proj_kernel<<<(Bk * Sk) / 16, 256, 0, stream>>>(outf16, WoH, bo, finalOut);
}
